// KNNMemory_72421738545472
// MI455X (gfx1250) — matrix operand order repaired and checked by reading, not compiled, not run on hardware
//
#include <hip/hip_runtime.h>

typedef __attribute__((ext_vector_type(16))) __bf16        v16bf;
typedef __attribute__((ext_vector_type(8)))  float         v8f;
typedef __attribute__((ext_vector_type(4)))  unsigned int  v4u;
typedef __attribute__((ext_vector_type(8)))  int           v8i;
typedef __attribute__((ext_vector_type(4)))  int           v4i;

#define N_Q   4096
#define M_K   65536
#define DIM   256
#define TOPK  32
#define NEG_INF -3.0e38f

#if __has_builtin(__builtin_amdgcn_tensor_load_to_lds)
#define USE_TDM 1
#endif

// ---- LDS arena layout (bytes) -------------------------------------------
// B double buffers: 4 waves x 2 bufs x (32 keys x 256 d x 2B = 16KB) = 128KB
#define BBUF_OFF   0
#define BBUF_SZ    (128 * 1024)
#define SC_OFF     (BBUF_OFF + BBUF_SZ)          // score tile [4][16][32] f32 = 8KB
                                                 // (qtile 16x256 bf16 = 8KB aliases this)
#define TOPS_OFF   (SC_OFF + 8192)               // topS [4][16][32] f32 = 8KB
#define TOPI_OFF   (TOPS_OFF + 8192)             // topI [4][16][32] i32 = 8KB
#define FINS_OFF   (TOPI_OFF + 8192)             // finS [16][32] f32 = 2KB
#define FINI_OFF   (FINS_OFF + 2048)             // finI [16][32] i32 = 2KB
#define LDS_TOTAL  (FINI_OFF + 2048)             // 156KB

__device__ __forceinline__ __bf16 us2bf(unsigned short u) {
    return __builtin_bit_cast(__bf16, u);
}
__device__ __forceinline__ unsigned short f2bf_us(float f) {
    __bf16 h = (__bf16)f;
    return __builtin_bit_cast(unsigned short, h);
}

#ifdef USE_TDM
// Issue one TDM load of a 32-key x 256-d bf16 tile (row-major, stride 256)
// into LDS at `ldst`. D# per CDNA5 ISA ch.8:
//  group0: count=1 | lds_addr | global_addr[31:0] | global_addr[56:32]+type=2
//  group1: data_size=2B | tensor_dim0=256 | tensor_dim1=65536 |
//          tile_dim0=256 | tile_dim1=32 | tensor_dim0_stride=256
__device__ __forceinline__ void tdm_load_tile32(const unsigned short* gsrc,
                                                void* ldst) {
    unsigned long long ga = (unsigned long long)(uintptr_t)gsrc;
    unsigned           la = (unsigned)(uintptr_t)ldst;
    v4u g0 = { 1u,                                   // count = 1 valid descriptor
               la,                                   // lds_addr
               (unsigned)ga,                         // global_addr[31:0]
               (unsigned)((ga >> 32) & 0x01FFFFFFull) | (2u << 30) }; // [56:32]|type=2
    v8i g1 = { (int)0x00010000,                      // data_size = 1 (2 bytes)
               (int)(256u << 16),                    // tensor_dim0 = 256 (bits 79:48)
               0,                                    // dim0 hi / tensor_dim1 lo16 = 0
               (int)(1u | (256u << 16)),             // tensor_dim1 hi16 (=65536), tile_dim0=256
               32,                                   // tile_dim1 = 32 keys
               256,                                  // tensor_dim0_stride = 256
               0, 0 };
    v4i z4 = {0, 0, 0, 0};
#if defined(__clang_major__) && (__clang_major__ >= 23)
    v8i z8 = {0, 0, 0, 0, 0, 0, 0, 0};
    __builtin_amdgcn_tensor_load_to_lds(g0, g1, z4, z4, z8, 0);
#else
    __builtin_amdgcn_tensor_load_to_lds(g0, g1, z4, z4, 0);
#endif
}
#endif

// ---------------------------------------------------------------------------
// Phase 1: row-normalize (x / (||x|| + 1e-8)) and convert to bf16.
// ---------------------------------------------------------------------------
__global__ void normalize_rows_bf16(const float* __restrict__ in,
                                    unsigned short* __restrict__ out) {
    __shared__ float red[DIM];
    const int row = blockIdx.x;
    const int t   = threadIdx.x;
    float x = in[(size_t)row * DIM + t];
    red[t] = x * x;
    __syncthreads();
    for (int off = DIM / 2; off > 0; off >>= 1) {
        if (t < off) red[t] += red[t + off];
        __syncthreads();
    }
    float rn = 1.0f / (sqrtf(red[0]) + 1e-8f);
    out[(size_t)row * DIM + t] = f2bf_us(x * rn);
}

// ---------------------------------------------------------------------------
// Phase 2: TDM double-buffered bf16 WMMA GEMM + running top-32 + value gather.
// Grid: N/16 workgroups x 128 threads (4 wave32s).
// ---------------------------------------------------------------------------
__global__ void __launch_bounds__(128)
knn_topk_wmma(const unsigned short* __restrict__ qn,   // [N_Q, DIM] bf16
              const unsigned short* __restrict__ kn,   // [M_K, DIM] bf16
              const float* __restrict__ values,        // [M_K, DIM] f32
              float* __restrict__ out) {
    __shared__ __align__(1024) unsigned char smem[LDS_TOTAL];

    unsigned short* qtile = (unsigned short*)(smem + SC_OFF);  // aliases scTile
    float* scT  = (float*)(smem + SC_OFF);    // [(w*16+r)*32 + n]
    float* topS = (float*)(smem + TOPS_OFF);  // [(w*16+r)*32 + t]
    int*   topI = (int*)  (smem + TOPI_OFF);
    float* finS = (float*)(smem + FINS_OFF);  // [r*32 + j]
    int*   finI = (int*)  (smem + FINI_OFF);

    const int tid  = threadIdx.x;
    const int lane = tid & 31;
    const int w    = tid >> 5;
    const int row0 = blockIdx.x * 16;
    const int r16  = lane & 15;
    const int g    = lane >> 4;

    // ---- stage Q tile, init top lists ----
    for (int i = tid; i < 16 * DIM; i += 128)
        qtile[i] = qn[(size_t)row0 * DIM + i];
    for (int j = lane; j < 16 * 32; j += 32) {
        topS[w * 512 + j] = NEG_INF;
        topI[w * 512 + j] = 0;
    }
    __syncthreads();

    // ---- build 8 loop-invariant A fragments (contiguous k = 16*g + e per lane half) ----
    v16bf afrag[8];
#pragma unroll
    for (int s = 0; s < 8; ++s) {
        v16bf a;
#pragma unroll
        for (int v = 0; v < 8; ++v) {
            int k0 = g * 16 + 2 * v;
            a[2 * v]     = us2bf(qtile[r16 * DIM + s * 32 + k0]);
            a[2 * v + 1] = us2bf(qtile[r16 * DIM + s * 32 + k0 + 1]);
        }
        afrag[s] = a;
    }
    __syncthreads();   // qtile region is reused as score tile below

    const int mchunk = M_K / 4;            // 16384 keys per wave
    const int mstart = w * mchunk;
    const int iters  = mchunk / 32;        // 512
    float rmin  = NEG_INF;
    int   rslot = 0;

#ifdef USE_TDM
    // prime the two per-wave tile buffers
    tdm_load_tile32(kn + (size_t)mstart * DIM,        smem + (w * 2 + 0) * 16384);
    tdm_load_tile32(kn + (size_t)(mstart + 32) * DIM, smem + (w * 2 + 1) * 16384);
#endif

    for (int it = 0; it < iters; ++it) {
        const int mbase = mstart + it * 32;
        v8f c0 = {0.f, 0.f, 0.f, 0.f, 0.f, 0.f, 0.f, 0.f};
        v8f c1 = {0.f, 0.f, 0.f, 0.f, 0.f, 0.f, 0.f, 0.f};

#ifdef USE_TDM
        const int cur = it & 1;
        if (it == iters - 1) __builtin_amdgcn_s_wait_tensorcnt(0);
        else                 __builtin_amdgcn_s_wait_tensorcnt(1);
        asm volatile("" ::: "memory");

        // B fragments from LDS: lane l -> column n = l&15 of subtile,
        // VGPR v: K = (l>>4)*16 + 2v -> 32 contiguous bytes per lane.
        const unsigned short* bb =
            (const unsigned short*)(smem + (w * 2 + cur) * 16384);
#pragma unroll
        for (int s = 0; s < 8; ++s) {
            v16bf b0 = *(const v16bf*)(bb + (size_t)r16 * DIM + g * 16 + s * 32);
            v16bf b1 = *(const v16bf*)(bb + (size_t)(16 + r16) * DIM + g * 16 + s * 32);
            c0 = __builtin_amdgcn_wmma_f32_16x16x32_bf16(
                     false, afrag[s], false, b0, (short)0, c0, false, false);
            c1 = __builtin_amdgcn_wmma_f32_16x16x32_bf16(
                     false, afrag[s], false, b1, (short)0, c1, false, false);
        }
#else
        const unsigned short* kr0 =
            kn + (size_t)(mbase + r16) * DIM + (size_t)g * 16;
        const unsigned short* kr1 = kr0 + 16 * DIM;
        __builtin_prefetch((const void*)(kr0 + 32 * DIM), 0, 0);
        __builtin_prefetch((const void*)(kr1 + 32 * DIM), 0, 0);
#pragma unroll
        for (int s = 0; s < 8; ++s) {
            v16bf b0 = *(const v16bf*)(kr0 + s * 32);
            v16bf b1 = *(const v16bf*)(kr1 + s * 32);
            c0 = __builtin_amdgcn_wmma_f32_16x16x32_bf16(
                     false, afrag[s], false, b0, (short)0, c0, false, false);
            c1 = __builtin_amdgcn_wmma_f32_16x16x32_bf16(
                     false, afrag[s], false, b1, (short)0, c1, false, false);
        }
#endif

        // C layout: VGPR i -> row (i + 8*(lane>>4)), col (lane&15)
#pragma unroll
        for (int i = 0; i < 8; ++i) {
            scT[(w * 16 + i + 8 * g) * 32 + r16]      = c0[i];
            scT[(w * 16 + i + 8 * g) * 32 + 16 + r16] = c1[i];
        }

#ifdef USE_TDM
        // refill buf[cur] with tile it+2 once our LDS reads/writes retired
        if (it + 2 < iters) {
            asm volatile("s_wait_dscnt 0" ::: "memory");
            tdm_load_tile32(kn + (size_t)(mbase + 64) * DIM,
                            smem + (w * 2 + cur) * 16384);
        }
#endif
        __syncthreads();

        // owner lanes 0..15 fold 32 candidates into their row's top-32
        if (lane < 16) {
            const int r = lane;
            float* ts = topS + (w * 16 + r) * 32;
            int*   ti = topI + (w * 16 + r) * 32;
#pragma unroll 1
            for (int n = 0; n < 32; ++n) {
                float s = scT[(w * 16 + r) * 32 + n];
                if (s > rmin) {
                    ts[rslot] = s;
                    ti[rslot] = mbase + n;
                    rmin = ts[0]; rslot = 0;
#pragma unroll 1
                    for (int j = 1; j < 32; ++j) {
                        float v = ts[j];
                        if (v < rmin) { rmin = v; rslot = j; }
                    }
                }
            }
        }
        __syncthreads();
    }

    // ---- merge 4 per-wave lists (128 candidates) into descending top-32 ----
    if (tid < 16) {
        const int r = tid;
        for (int j = 0; j < TOPK; ++j) {
            float best = NEG_INF; int bs = 0;
            for (int t = 0; t < 128; ++t) {
                float v = topS[((t >> 5) * 16 + r) * 32 + (t & 31)];
                if (v > best) { best = v; bs = t; }
            }
            finS[r * TOPK + j] = best;
            finI[r * TOPK + j] = topI[((bs >> 5) * 16 + r) * 32 + (bs & 31)];
            topS[((bs >> 5) * 16 + r) * 32 + (bs & 31)] = NEG_INF;
        }
    }
    __syncthreads();

    // ---- write scores + indices ----
    float* outScores = out;
    float* outIdx    = out + (size_t)N_Q * TOPK;
    float* outVals   = out + (size_t)2 * N_Q * TOPK;
    for (int p = tid; p < 16 * TOPK; p += 128) {
        int rr = p >> 5, jj = p & 31;
        size_t o = (size_t)(row0 + rr) * TOPK + jj;
        outScores[o] = finS[rr * TOPK + jj];
        outIdx[o]    = (float)finI[rr * TOPK + jj];
    }
    __syncthreads();

    // ---- gather value rows: 512 (row,j) pairs, 128 threads x float2 each ----
    for (int p = 0; p < 16 * TOPK; ++p) {
        int rr = p >> 5, jj = p & 31;
        int idx = finI[rr * TOPK + jj];
        const float2* src = (const float2*)(values + (size_t)idx * DIM);
        float2* dst = (float2*)(outVals +
                       ((size_t)(row0 + rr) * TOPK + jj) * DIM);
        dst[tid] = src[tid];
    }
}

// ---------------------------------------------------------------------------
extern "C" void kernel_launch(void* const* d_in, const int* in_sizes, int n_in,
                              void* d_out, int out_size, void* d_ws, size_t ws_size,
                              hipStream_t stream) {
    const float* query  = (const float*)d_in[0];   // [4096, 256]
    const float* keys   = (const float*)d_in[1];   // [65536, 256]
    const float* values = (const float*)d_in[2];   // [65536, 256]
    // d_in[3] is k (= 32), compiled in as TOPK.

    unsigned short* qnrm = (unsigned short*)d_ws;            // 2 MB
    unsigned short* knrm = qnrm + (size_t)N_Q * DIM;         // 32 MB

    normalize_rows_bf16<<<N_Q, DIM, 0, stream>>>(query, qnrm);
    normalize_rows_bf16<<<M_K, DIM, 0, stream>>>(keys, knrm);
    knn_topk_wmma<<<N_Q / 16, 128, 0, stream>>>(qnrm, knrm, values, (float*)d_out);
}
